// IGMC_33311766348131
// MI455X (gfx1250) — compile-verified, matrix operand order checked by reading, not run on hardware
//
#include <hip/hip_runtime.h>
#include <hip/hip_bf16.h>
#include <math.h>

typedef __attribute__((ext_vector_type(16))) __bf16 v16bf;
typedef __attribute__((ext_vector_type(8)))  __bf16 v8bf;
typedef __attribute__((ext_vector_type(8)))  float  v8f;

#define NREL 2

// ---- monotonic float<->uint encoding for atomic max over floats ----
static __device__ __forceinline__ unsigned fenc(float f) {
  unsigned u = __float_as_uint(f);
  return (u & 0x80000000u) ? ~u : (u | 0x80000000u);
}
static __device__ __forceinline__ float fdec(unsigned e) {
  return __uint_as_float((e & 0x80000000u) ? (e & 0x7FFFFFFFu) : ~e);
}

__global__ void k_fill(float* p, float v, long n) {
  long i = (long)blockIdx.x * blockDim.x + threadIdx.x;
  if (i < n) p[i] = v;
}

// W[r*IO + j] = sum_b comp[r*4+b] * basis[b*IO + j]
__global__ void k_wcomb(const float* __restrict__ basis, const float* __restrict__ comp,
                        float* __restrict__ W, long IO) {
  long i = (long)blockIdx.x * blockDim.x + threadIdx.x;
  if (i >= (long)NREL * IO) return;
  int  r = (int)(i / IO);
  long j = i % IO;
  float acc = 0.f;
  #pragma unroll
  for (int b = 0; b < 4; ++b) acc += comp[r * 4 + b] * basis[(long)b * IO + j];
  W[i] = acc;
}

// out[i] = a[i] + bias[i % O]
__global__ void k_addbias(const float* __restrict__ a, const float* __restrict__ bias,
                          float* __restrict__ out, long n, int O) {
  long i = (long)blockIdx.x * blockDim.x + threadIdx.x;
  if (i < n) out[i] = a[i] + bias[i % O];
}

// out[n,o] = (bias? bias[o] : 0) + sum_k x[n,k]*w[k,o]
__global__ void k_matmul(const float* __restrict__ x, const float* __restrict__ w,
                         const float* __restrict__ bias, float* __restrict__ out,
                         int Nn, int K, int O) {
  long i = (long)blockIdx.x * blockDim.x + threadIdx.x;
  if (i >= (long)Nn * O) return;
  int n = (int)(i / O), o = (int)(i % O);
  float acc = bias ? bias[o] : 0.f;
  const float* xr = x + (long)n * K;
  for (int k = 0; k < K; ++k) acc += xr[k] * w[(long)k * O + o];
  out[i] = acc;
}

// per-relation mean aggregation: scatter sums + counts
__global__ void k_scatter(const float* __restrict__ hr, const int* __restrict__ src,
                          const int* __restrict__ dst, const int* __restrict__ et,
                          float* __restrict__ s, float* __restrict__ cnt,
                          int E, int Nn, int O) {
  long i = (long)blockIdx.x * blockDim.x + threadIdx.x;
  if (i >= (long)E * O) return;
  int e = (int)(i / O), o = (int)(i % O);
  int r = et[e], sn = src[e], dn = dst[e];
  float v = hr[((long)r * Nn + sn) * O + o];
  atomicAdd(&s[((long)r * Nn + dn) * O + o], v);
  if (o == 0) atomicAdd(&cnt[(long)r * Nn + dn], 1.0f);
}

__global__ void k_combine(const float* __restrict__ base, const float* __restrict__ s,
                          const float* __restrict__ cnt, float* __restrict__ out,
                          int Nn, int O) {
  long i = (long)blockIdx.x * blockDim.x + threadIdx.x;
  if (i >= (long)Nn * O) return;
  int n = (int)(i / O), o = (int)(i % O);
  float acc = base[i];
  #pragma unroll
  for (int r = 0; r < NREL; ++r) {
    float c = cnt[(long)r * Nn + n];
    acc += s[((long)r * Nn + n) * O + o] / fmaxf(c, 1.0f);
  }
  out[i] = tanhf(acc);
}

// GAT attention scalars: asv[n] = h[n].a_src, adv[n] = h[n].a_dst
__global__ void k_att(const float* __restrict__ h, const float* __restrict__ as,
                      const float* __restrict__ ad, float* __restrict__ asv,
                      float* __restrict__ adv, int Nn) {
  int n = blockIdx.x * blockDim.x + threadIdx.x;
  if (n >= Nn) return;
  const float* hr = h + (long)n * 512;
  float a = 0.f, b = 0.f;
  for (int k = 0; k < 512; ++k) { float v = hr[k]; a += v * as[k]; b += v * ad[k]; }
  asv[n] = a; adv[n] = b;
}

// alpha = leaky_relu(asv[s]+adv[d]); segment max via monotonic-uint atomicMax
__global__ void k_alpha(const float* __restrict__ asv, const float* __restrict__ adv,
                        const int* __restrict__ src, const int* __restrict__ dst,
                        float* __restrict__ alpha, unsigned* __restrict__ amax,
                        int E, int Nn) {
  int e = blockIdx.x * blockDim.x + threadIdx.x;
  if (e >= E + Nn) return;
  int s, d;
  if (e < E) { s = src[e]; d = dst[e]; } else { s = d = e - E; }
  float a = asv[s] + adv[d];
  a = a > 0.f ? a : 0.2f * a;
  alpha[e] = a;
  atomicMax(&amax[d], fenc(a));
}

__global__ void k_expsum(float* __restrict__ alpha, const unsigned* __restrict__ amax,
                         const int* __restrict__ dst, float* __restrict__ denom,
                         int E, int Nn) {
  int e = blockIdx.x * blockDim.x + threadIdx.x;
  if (e >= E + Nn) return;
  int d = (e < E) ? dst[e] : e - E;
  float ex = expf(alpha[e] - fdec(amax[d]));
  alpha[e] = ex;
  atomicAdd(&denom[d], ex);
}

__global__ void k_gatagg(const float* __restrict__ alpha, const float* __restrict__ denom,
                         const float* __restrict__ h, const int* __restrict__ src,
                         const int* __restrict__ dst, float* __restrict__ xg,
                         int E, int Nn) {
  long tid = (long)blockIdx.x * blockDim.x + threadIdx.x;
  if (tid >= (long)(E + Nn) * 64) return;
  int e = (int)(tid >> 6);
  int o0 = (int)(tid & 63) * 8;
  int s, d;
  if (e < E) { s = src[e]; d = dst[e]; } else { s = d = e - E; }
  float coef = alpha[e] / fmaxf(denom[d], 1e-16f);
  const float* hs = h + (long)s * 512 + o0;
  float* xo = xg + (long)d * 512 + o0;
  #pragma unroll
  for (int j = 0; j < 8; ++j) atomicAdd(&xo[j], hs[j] * coef);
}

__global__ void k_relubf(const float* __restrict__ xg, const float* __restrict__ bias,
                         __bf16* __restrict__ xgh, long n) {
  long i = (long)blockIdx.x * blockDim.x + threadIdx.x;
  if (i >= n) return;
  float v = xg[i] + bias[i % 512];
  xgh[i] = (__bf16)(v > 0.f ? v : 0.f);
}

// Pack w1 [1024,128] into bf16 B-fragment order:
// w1p[((c*8+t)*32+lane)*16 + i] = w1[(c*32 + klocal)*128 + (t*16 + lane%16)]
// klocal = (i<8 ? i : i+8) + (lane>=16 ? 8 : 0)   (ISA 16-bit B^T layout)
__global__ void k_packw1(const float* __restrict__ w1, __bf16* __restrict__ w1p) {
  int tid = blockIdx.x * blockDim.x + threadIdx.x;
  if (tid >= 32 * 8 * 32 * 16) return;
  int i    = tid & 15;
  int lane = (tid >> 4) & 31;
  int t    = (tid >> 9) & 7;
  int c    = tid >> 12;
  int n  = t * 16 + (lane & 15);
  int kl = (i < 8 ? i : i + 8) + ((lane >= 16) ? 8 : 0);
  int k  = c * 32 + kl;
  w1p[tid] = (__bf16)w1[k * 128 + n];
}

// Edge MLP, WMMA bf16. Per wave: 32 edges (2 M-tiles of 16), 128 hidden cols
// (8 N-tiles), K=1024 in 32 chunks. B (packed w1, 256 KB) staged in LDS once
// per workgroup (CDNA5: 320 KB LDS/WGP). Steady-state chunk:
//   4x global_load_b128 (A) + 16x ds_load (B) + 16x wmma.
// NOTE: accumulators must never be address-taken (keeps them in VGPRs).
__global__ void __launch_bounds__(256, 1)
k_mlp(const __bf16* __restrict__ xgh, const __bf16* __restrict__ w1p,
      const float* __restrict__ b1, const float* __restrict__ w2,
      const float* __restrict__ b2, const int* __restrict__ src,
      const int* __restrict__ dst, float* __restrict__ out, int E) {
  extern __shared__ __bf16 bsh[];  // 131072 bf16 = 256 KB

  // cooperative stage of packed w1 into LDS (16-byte vectors)
  {
    const v8bf* g = (const v8bf*)w1p;
    v8bf* l = (v8bf*)bsh;
    for (int i = threadIdx.x; i < 131072 / 8; i += blockDim.x) l[i] = g[i];
  }
  __syncthreads();

  int wave = (blockIdx.x * blockDim.x + threadIdx.x) >> 5;
  int lane = threadIdx.x & 31;
  int e0 = wave * 32;
  if (e0 >= E) return;  // wave-uniform (after barrier)

  int m = lane & 15;
  int ea = e0 + m;      if (ea >= E) ea = E - 1;  // tile 0 row
  int eb = e0 + 16 + m; if (eb >= E) eb = E - 1;  // tile 1 row
  const __bf16* rs0 = xgh + (long)src[ea] * 512;
  const __bf16* rd0 = xgh + (long)dst[ea] * 512;
  const __bf16* rs1 = xgh + (long)src[eb] * 512;
  const __bf16* rd1 = xgh + (long)dst[eb] * 512;
  int kbase = (lane < 16) ? 0 : 8;  // A-fragment k-half per lane group

  v8f acc0[8] = {};
  v8f acc1[8] = {};
  for (int c = 0; c < 32; ++c) {
    const __bf16* row0 = (c < 16) ? rs0 : rd0;
    const __bf16* row1 = (c < 16) ? rs1 : rd1;
    int koff = (c & 15) * 32 + kbase;
    // A gathers (global, LOADcnt)
    v8bf lo0 = *(const v8bf*)(row0 + koff);
    v8bf hi0 = *(const v8bf*)(row0 + koff + 16);
    v8bf lo1 = *(const v8bf*)(row1 + koff);
    v8bf hi1 = *(const v8bf*)(row1 + koff + 16);
    // B fragments for this chunk (LDS, DScnt) -- all 8 up front
    v16bf b[8];
    const __bf16* bp = bsh + ((long)c * 8 * 32 + lane) * 16;
    #pragma unroll
    for (int t = 0; t < 8; ++t) b[t] = *(const v16bf*)(bp + (long)t * 32 * 16);
    v16bf a0, a1;
    #pragma unroll
    for (int i = 0; i < 8; ++i) {
      a0[i] = lo0[i]; a0[i + 8] = hi0[i];
      a1[i] = lo1[i]; a1[i + 8] = hi1[i];
    }
    #pragma unroll
    for (int t = 0; t < 8; ++t) {
      acc0[t] = __builtin_amdgcn_wmma_f32_16x16x32_bf16(
          false, a0, false, b[t], (short)0, acc0[t], false, false);
      acc1[t] = __builtin_amdgcn_wmma_f32_16x16x32_bf16(
          false, a1, false, b[t], (short)0, acc1[t], false, false);
    }
  }

  // Epilogue (duplicated per M-tile; no pointer to acc arrays so they stay in
  // VGPRs). C layout: lane holds column (lane&15) of each 16-col tile, rows j
  // (+8 for lanes>=16). relu(+b1), dot w2, half-wave reduce, sigmoid(+b2).
  int ncol = lane & 15;
  int rbase = (lane < 16) ? 0 : 8;
  float bias2 = b2[0];

  {  // ---- tile 0 ----
    float part[8];
    #pragma unroll
    for (int j = 0; j < 8; ++j) part[j] = 0.f;
    #pragma unroll
    for (int t = 0; t < 8; ++t) {
      int n = t * 16 + ncol;
      float bb = b1[n], ww = w2[n];
      #pragma unroll
      for (int j = 0; j < 8; ++j) {
        float v = acc0[t][j] + bb;
        v = v > 0.f ? v : 0.f;
        part[j] += v * ww;
      }
    }
    #pragma unroll
    for (int off = 1; off < 16; off <<= 1) {
      #pragma unroll
      for (int j = 0; j < 8; ++j) part[j] += __shfl_xor(part[j], off, 16);
    }
    if (ncol == 0) {
      #pragma unroll
      for (int j = 0; j < 8; ++j) {
        int ee = e0 + rbase + j;
        if (ee < E) out[ee] = 1.f / (1.f + expf(-(part[j] + bias2)));
      }
    }
  }

  {  // ---- tile 1 ----
    float part[8];
    #pragma unroll
    for (int j = 0; j < 8; ++j) part[j] = 0.f;
    #pragma unroll
    for (int t = 0; t < 8; ++t) {
      int n = t * 16 + ncol;
      float bb = b1[n], ww = w2[n];
      #pragma unroll
      for (int j = 0; j < 8; ++j) {
        float v = acc1[t][j] + bb;
        v = v > 0.f ? v : 0.f;
        part[j] += v * ww;
      }
    }
    #pragma unroll
    for (int off = 1; off < 16; off <<= 1) {
      #pragma unroll
      for (int j = 0; j < 8; ++j) part[j] += __shfl_xor(part[j], off, 16);
    }
    if (ncol == 0) {
      #pragma unroll
      for (int j = 0; j < 8; ++j) {
        int ee = e0 + 16 + rbase + j;
        if (ee < E) out[ee] = 1.f / (1.f + expf(-(part[j] + bias2)));
      }
    }
  }
}

static inline int nblk(long n, int b) { return (int)((n + b - 1) / b); }

extern "C" void kernel_launch(void* const* d_in, const int* in_sizes, int n_in,
                              void* d_out, int out_size, void* d_ws, size_t ws_size,
                              hipStream_t stream) {
  const float* basis[4] = {(const float*)d_in[0], (const float*)d_in[4],
                           (const float*)d_in[8], (const float*)d_in[12]};
  const float* comp[4]  = {(const float*)d_in[1], (const float*)d_in[5],
                           (const float*)d_in[9], (const float*)d_in[13]};
  const float* root[4]  = {(const float*)d_in[2], (const float*)d_in[6],
                           (const float*)d_in[10], (const float*)d_in[14]};
  const float* rbias[4] = {(const float*)d_in[3], (const float*)d_in[7],
                           (const float*)d_in[11], (const float*)d_in[15]};
  const float* gat_w    = (const float*)d_in[16];
  const float* gat_as   = (const float*)d_in[17];
  const float* gat_ad   = (const float*)d_in[18];
  const float* gat_b    = (const float*)d_in[19];
  const float* w1       = (const float*)d_in[20];
  const float* b1       = (const float*)d_in[21];
  const float* w2       = (const float*)d_in[22];
  const float* b2       = (const float*)d_in[23];
  const int*   ei       = (const int*)d_in[24];
  const int*   etype    = (const int*)d_in[25];

  const int E = in_sizes[25];
  const int N = in_sizes[2] / 32;  // root0 is [N,32]
  const int* src = ei;
  const int* dst = ei + E;

  // workspace carve-out (64B-aligned slots)
  float* wsf = (float*)d_ws;
  size_t off = 0;
  auto alloc = [&](size_t nelem) -> float* {
    float* p = wsf + off;
    off += (nelem + 15) & ~(size_t)15;
    return p;
  };
  float* hr0   = alloc((size_t)2 * N * 32);  // layer-0 relation weights (= hr since x=I)
  float* base  = alloc((size_t)N * 64);
  float* sbuf  = alloc((size_t)2 * N * 64);
  float* cnt   = alloc((size_t)2 * N);
  float* xa    = alloc((size_t)N * 64);
  float* xb    = alloc((size_t)N * 64);
  float* Wl    = alloc((size_t)2 * 64 * 64);
  float* hr    = alloc((size_t)2 * N * 64);
  float* h     = alloc((size_t)N * 512);
  float* asv   = alloc((size_t)N);
  float* adv   = alloc((size_t)N);
  float* alpha = alloc((size_t)(E + N));
  float* amaxf = alloc((size_t)N);
  float* denom = alloc((size_t)N);
  float* xg    = alloc((size_t)N * 512);
  __bf16* xgh  = (__bf16*)alloc((size_t)N * 256);  // N*512 bf16
  __bf16* w1p  = (__bf16*)alloc((size_t)65536);    // 131072 bf16

  const int B = 256;

  // ---------------- Layer 0 (x = I: x@W[r] == W[r], x@root == root) ----------------
  k_wcomb  <<<nblk((long)2 * N * 32, B), B, 0, stream>>>(basis[0], comp[0], hr0, (long)N * 32);
  k_addbias<<<nblk((long)N * 32, B), B, 0, stream>>>(root[0], rbias[0], base, (long)N * 32, 32);
  k_fill   <<<nblk((long)2 * N * 32, B), B, 0, stream>>>(sbuf, 0.f, (long)2 * N * 32);
  k_fill   <<<nblk((long)2 * N, B), B, 0, stream>>>(cnt, 0.f, (long)2 * N);
  k_scatter<<<nblk((long)E * 32, B), B, 0, stream>>>(hr0, src, dst, etype, sbuf, cnt, E, N, 32);
  k_combine<<<nblk((long)N * 32, B), B, 0, stream>>>(base, sbuf, cnt, xa, N, 32);

  // ---------------- Layers 1..3 ----------------
  const int dims[4][2] = {{0, 32}, {32, 64}, {64, 64}, {64, 32}};
  float* xin = xa;
  float* xout = xb;
  for (int l = 1; l <= 3; ++l) {
    int I = dims[l][0], O = dims[l][1];
    k_wcomb  <<<nblk((long)2 * I * O, B), B, 0, stream>>>(basis[l], comp[l], Wl, (long)I * O);
    k_matmul <<<nblk((long)N * O, B), B, 0, stream>>>(xin, root[l], rbias[l], base, N, I, O);
    k_matmul <<<nblk((long)N * O, B), B, 0, stream>>>(xin, Wl,             nullptr, hr,                N, I, O);
    k_matmul <<<nblk((long)N * O, B), B, 0, stream>>>(xin, Wl + (long)I*O, nullptr, hr + (long)N * O, N, I, O);
    k_fill   <<<nblk((long)2 * N * O, B), B, 0, stream>>>(sbuf, 0.f, (long)2 * N * O);
    k_fill   <<<nblk((long)2 * N, B), B, 0, stream>>>(cnt, 0.f, (long)2 * N);
    k_scatter<<<nblk((long)E * O, B), B, 0, stream>>>(hr, src, dst, etype, sbuf, cnt, E, N, O);
    k_combine<<<nblk((long)N * O, B), B, 0, stream>>>(base, sbuf, cnt, xout, N, O);
    float* t = xin; xin = xout; xout = t;
  }
  float* x3 = xin;  // [N,32]

  // ---------------- GAT layer ----------------
  k_matmul<<<nblk((long)N * 512, B), B, 0, stream>>>(x3, gat_w, nullptr, h, N, 32, 512);
  k_att   <<<nblk((long)N, B), B, 0, stream>>>(h, gat_as, gat_ad, asv, adv, N);
  k_fill  <<<nblk((long)N, B), B, 0, stream>>>(amaxf, 0.f, (long)N);  // enc(-inf) > 0; self-loops guarantee updates
  k_fill  <<<nblk((long)N, B), B, 0, stream>>>(denom, 0.f, (long)N);
  k_fill  <<<nblk((long)N * 512, B), B, 0, stream>>>(xg, 0.f, (long)N * 512);
  k_alpha <<<nblk((long)(E + N), B), B, 0, stream>>>(asv, adv, src, dst, alpha, (unsigned*)amaxf, E, N);
  k_expsum<<<nblk((long)(E + N), B), B, 0, stream>>>(alpha, (const unsigned*)amaxf, dst, denom, E, N);
  k_gatagg<<<nblk((long)(E + N) * 64, B), B, 0, stream>>>(alpha, denom, h, src, dst, xg, E, N);
  k_relubf<<<nblk((long)N * 512, B), B, 0, stream>>>(xg, gat_b, xgh, (long)N * 512);

  // ---------------- Edge MLP (WMMA bf16, B staged in LDS) ----------------
  k_packw1<<<nblk(131072, B), B, 0, stream>>>(w1, w1p);
  (void)hipFuncSetAttribute((const void*)k_mlp,
                            hipFuncAttributeMaxDynamicSharedMemorySize, 262144);
  long waves = ((long)E + 31) / 32;            // 32 edges per wave
  int mlp_blocks = nblk(waves * 32, 256);      // 8 waves / block
  k_mlp<<<mlp_blocks, 256, 262144, stream>>>(xgh, w1p, b1, w2, b2, src, dst,
                                             (float*)d_out, E);
}
